// Self_Match_38053410242637
// MI455X (gfx1250) — matrix operand order repaired and checked by reading, not compiled, not run on hardware
//
#include <hip/hip_runtime.h>
#include <hip/hip_bf16.h>

typedef __attribute__((ext_vector_type(16))) __bf16 v16bf;
typedef __attribute__((ext_vector_type(8)))  float  v8f;

#define NSAMP 4
#define HP 64
#define L 4096           // HP*HP patches per sample
#define KRAW 75          // 3*5*5
#define KP 96            // padded K (3 steps of 32)
#define KL 104           // LDS row stride (multiple of 8 -> 16B aligned rows)
#define BM 128
#define BN 128

union FragU { uint4 u[2]; v16bf v; };

// ---------------------------------------------------------------- kernel 0
__global__ void init_kernel(float* maxnsq) {
    if (threadIdx.x < NSAMP) maxnsq[threadIdx.x] = 0.0f;
}

// ---------------------------------------------------------------- kernel 1
// One wave (32 lanes) per patch: gather 75 fp32 values, emit bf16 padded to 96,
// reduce squared norm, atomic-max per sample.
__global__ void patch_kernel(const float* __restrict__ x,
                             __bf16* __restrict__ P,
                             float* __restrict__ maxnsq) {
    const int gid  = blockIdx.x * 8 + (threadIdx.x >> 5);   // patch id, 0..16383
    const int lane = threadIdx.x & 31;
    const int n = gid >> 12;
    const int p = gid & (L - 1);
    const int py = p >> 6, px = p & 63;

    float ssq = 0.0f;
#pragma unroll
    for (int t = 0; t < 3; ++t) {
        const int kk = lane + t * 32;               // 0..95
        float v = 0.0f;
        if (kk < KRAW) {
            const int c  = kk / 25;
            const int r  = kk - c * 25;
            const int ki = r / 5;
            const int kj = r - ki * 5;
            v = x[(((size_t)(n * 3 + c) * 320) + py * 5 + ki) * 320 + px * 5 + kj];
        }
        P[((size_t)(n * L + p)) * KP + kk] = (__bf16)v;
        ssq += v * v;
    }
#pragma unroll
    for (int off = 16; off > 0; off >>= 1)
        ssq += __shfl_down(ssq, off, 32);
    if (lane == 0)
        atomicMax((unsigned int*)&maxnsq[n], __float_as_uint(ssq));  // ssq >= 0
}

// ---------------------------------------------------------------- kernel 2
__global__ void scale_kernel(const float* __restrict__ maxnsq,
                             float* __restrict__ scale) {
    if (threadIdx.x < NSAMP) scale[threadIdx.x] = rsqrtf(maxnsq[threadIdx.x]);
}

// ---------------------------------------------------------------- kernel 3
// Gram tile: out[n, p0+M, q0+N] = sigmoid(scale[n] * dot(P[p0+M], P[q0+N]))
// 256 threads = 8 waves arranged 2x4; each wave: 4 row-subtiles x 2 col-subtiles.
__global__ void __launch_bounds__(256)
gram_kernel(const __bf16* __restrict__ P,
            const float* __restrict__ scale,
            float* __restrict__ out) {
    __shared__ __bf16 lAs[BM * KL];
    __shared__ __bf16 lBs[BN * KL];

    const int n  = blockIdx.z;
    const int p0 = blockIdx.x * BM;
    const int q0 = blockIdx.y * BN;
    const __bf16* Pn = P + (size_t)n * L * KP;

    // Stage A (rows p0..p0+127) and B (rows q0..q0+127) panels into LDS.
    // 128 rows x 12 chunks of 8 bf16 (16B) = 1536 chunks per panel; 6 per thread.
#pragma unroll
    for (int j = 0; j < 6; ++j) {
        const int i   = threadIdx.x + j * 256;
        const int row = i / 12;
        const int col = (i % 12) * 8;
        uint4 va = *(const uint4*)(Pn + (size_t)(p0 + row) * KP + col);
        uint4 vb = *(const uint4*)(Pn + (size_t)(q0 + row) * KP + col);
        *(uint4*)&lAs[row * KL + col] = va;
        *(uint4*)&lBs[row * KL + col] = vb;
    }
    __syncthreads();

    const int wave = threadIdx.x >> 5;
    const int lane = threadIdx.x & 31;
    const int wr = wave >> 2;        // 0..1  -> 64-row slab
    const int wc = wave & 3;         // 0..3  -> 32-col slab
    const int half  = lane >> 4;     // 0/1
    const int lmod  = lane & 15;

    v8f acc[4][2];
#pragma unroll
    for (int i = 0; i < 4; ++i)
#pragma unroll
        for (int j = 0; j < 2; ++j)
            acc[i][j] = (v8f){};

#pragma unroll
    for (int ks = 0; ks < 3; ++ks) {
        const int k = ks * 32;

        // A fragments: 16-bit A 16x32 layout. Lane (half,lmod): M=lmod,
        // elems 0..7 -> K = k + half*8 + e ; elems 8..15 -> K = k + 16 + half*8 + e.
        FragU a[4];
#pragma unroll
        for (int i = 0; i < 4; ++i) {
            const int rowM = wr * 64 + i * 16 + lmod;
            const __bf16* src = &lAs[rowM * KL + k + half * 8];
            a[i].u[0] = *(const uint4*)(src);
            a[i].u[1] = *(const uint4*)(src + 16);
        }
        // B fragments: 16-bit B 32x16 layout. Lane: N=lmod, elems 0..7 ->
        // K = k + half*8 + e ; elems 8..15 -> K = k + 16 + half*8 + e.
        FragU b[2];
#pragma unroll
        for (int j = 0; j < 2; ++j) {
            const int colN = q0 ? (wc * 32 + j * 16 + lmod) : (wc * 32 + j * 16 + lmod);
            const __bf16* src = &lBs[(wc * 32 + j * 16 + lmod) * KL + k + half * 8];
            (void)colN;
            b[j].u[0] = *(const uint4*)(src);
            b[j].u[1] = *(const uint4*)(src + 16);
        }

#pragma unroll
        for (int i = 0; i < 4; ++i)
#pragma unroll
            for (int j = 0; j < 2; ++j)
                acc[i][j] = __builtin_amdgcn_wmma_f32_16x16x32_bf16(
                    false, a[i].v, false, b[j].v,
                    (short)0, acc[i][j], false, false);
    }

    // Epilogue: C/D layout -> element r of v8f in lane: M = r + half*8, N = lmod.
    const float s = scale[n];
#pragma unroll
    for (int i = 0; i < 4; ++i) {
#pragma unroll
        for (int j = 0; j < 2; ++j) {
            const int Nc = q0 + wc * 32 + j * 16 + lmod;
#pragma unroll
            for (int r = 0; r < 8; ++r) {
                const int Mr = p0 + wr * 64 + i * 16 + r + half * 8;
                const float v = acc[i][j][r] * s;
                const float sg = 1.0f / (1.0f + __expf(-v));
                out[((size_t)(n * L + Mr)) * L + Nc] = sg;
            }
        }
    }
}

// ---------------------------------------------------------------- launch
extern "C" void kernel_launch(void* const* d_in, const int* in_sizes, int n_in,
                              void* d_out, int out_size, void* d_ws, size_t ws_size,
                              hipStream_t stream) {
    const float* x = (const float*)d_in[0];
    float* out = (float*)d_out;

    // Workspace layout
    const size_t pBytes = (size_t)NSAMP * L * KP * sizeof(__bf16);  // 3 MB
    __bf16* P      = (__bf16*)d_ws;
    float*  maxnsq = (float*)((char*)d_ws + pBytes);
    float*  scl    = maxnsq + 8;

    init_kernel<<<1, 64, 0, stream>>>(maxnsq);
    patch_kernel<<<(NSAMP * L) / 8, 256, 0, stream>>>(x, P, maxnsq);
    scale_kernel<<<1, 64, 0, stream>>>(maxnsq, scl);

    dim3 grid(L / BM, L / BN, NSAMP);   // 32 x 32 x 4
    gram_kernel<<<grid, 256, 0, stream>>>(P, scl, out);
}
